// MoELayer_58884001628265
// MI455X (gfx1250) — compile-verified, matrix operand order checked by reading, not run on hardware
//
#include <hip/hip_runtime.h>
#include <hip/hip_bf16.h>

typedef __bf16 bf16;
typedef __attribute__((ext_vector_type(16))) __bf16 v16bf;
typedef __attribute__((ext_vector_type(8)))  __bf16 v8bf;
typedef __attribute__((ext_vector_type(2)))  __bf16 v2bf;
typedef __attribute__((ext_vector_type(8)))  float  v8f;
typedef __attribute__((ext_vector_type(4)))  float  v4f;
typedef __attribute__((ext_vector_type(4)))  int    v4i;

#define TOKENS 8192
#define DMODEL 2048
#define NEXP   8
#define FEXP   1024
#define FSH    2048

#define BM 128
#define BN 128
#define BK 64   // two 16x16x32 k-steps per LDS stage

#if __has_builtin(__builtin_amdgcn_global_load_async_to_lds_b128)
#define HAVE_ASYNC_LDS 1
typedef __attribute__((address_space(1))) v4i gv4i;  // global 128b unit
typedef __attribute__((address_space(3))) v4i lv4i;  // LDS 128b unit
#endif

__device__ __forceinline__ void wait_async_lds() {
#if defined(HAVE_ASYNC_LDS)
#if __has_builtin(__builtin_amdgcn_s_wait_asynccnt)
  __builtin_amdgcn_s_wait_asynccnt(0);
#else
  asm volatile("s_wait_asynccnt 0x0" ::: "memory");
#endif
#endif
}

// Copy one 128x64 bf16 A tile (already bf16 in ws) global -> LDS.
// Prefers the gfx1250 async DMA path (ASYNCcnt) over the VGPR round-trip.
__device__ __forceinline__ void stage_A_tile(const bf16* __restrict__ A,
                                             bf16 (*sA)[BK], int m0, int k0,
                                             int K, int tid) {
#pragma unroll
  for (int r = 0; r < 4; ++r) {
    int q = (r * 256 + tid) * 8;          // elem index in 128x64 tile
    int row = q >> 6, col = q & 63;
    const bf16* gsrc = &A[(size_t)(m0 + row) * K + k0 + col];
    bf16* ldst = &sA[row][col];
#if defined(HAVE_ASYNC_LDS)
    __builtin_amdgcn_global_load_async_to_lds_b128((gv4i*)gsrc, (lv4i*)ldst,
                                                   0, 0);
#else
    *(v8bf*)ldst = *(const v8bf*)gsrc;
#endif
  }
}

// Stage one 64x128 fp32 B tile -> LDS transposed [BN][BK] bf16, converting
// with packed bf16 pair stores (row pairs contiguous in the transposed tile).
__device__ __forceinline__ void stage_B_tile(const float* __restrict__ B,
                                             bf16 (*sB)[BK], int k0, int n0,
                                             int N, int tid) {
#pragma unroll
  for (int r = 0; r < 4; ++r) {
    int p = r * 256 + tid;                 // pair cell: 32 rowpairs x 32 col4
    int col4 = (p & 31) * 4;
    int row  = (p >> 5) * 2;
    size_t g0 = (size_t)(k0 + row) * N + n0 + col4;
    v4f a = *(const v4f*)&B[g0];
    v4f b = *(const v4f*)&B[g0 + N];
    v2bf pk;
    pk.x = (bf16)a.x; pk.y = (bf16)b.x; *(v2bf*)&sB[col4 + 0][row] = pk;
    pk.x = (bf16)a.y; pk.y = (bf16)b.y; *(v2bf*)&sB[col4 + 1][row] = pk;
    pk.x = (bf16)a.z; pk.y = (bf16)b.z; *(v2bf*)&sB[col4 + 2][row] = pk;
    pk.x = (bf16)a.w; pk.y = (bf16)b.w; *(v2bf*)&sB[col4 + 3][row] = pk;
  }
}

// ---------------------------------------------------------------------------
// fp32 -> bf16 conversion for h (done once)
// ---------------------------------------------------------------------------
__global__ void cvt_f32_to_bf16(const float* __restrict__ x,
                                bf16* __restrict__ y, int n) {
  int idx = (blockIdx.x * blockDim.x + threadIdx.x) * 4;
  if (idx + 3 < n) {
    v4f v = *(const v4f*)(x + idx);
    y[idx + 0] = (bf16)v.x;
    y[idx + 1] = (bf16)v.y;
    y[idx + 2] = (bf16)v.z;
    y[idx + 3] = (bf16)v.w;
  }
}

// ---------------------------------------------------------------------------
// Router: one wave32 per token (E=8 logits + shared gate logit), butterfly
// reduce, softmax + top-2 on lane 0.
// ---------------------------------------------------------------------------
__global__ __launch_bounds__(256)
void router_kernel(const float* __restrict__ h,      // [T, D]
                   const float* __restrict__ gate_w, // [D, E]
                   const float* __restrict__ wsg,    // [D, 1]
                   float* __restrict__ comb,         // [T, E]
                   float* __restrict__ gsh) {        // [T]
  const int lane = threadIdx.x & 31;
  const int wid  = threadIdx.x >> 5;
  const int t    = blockIdx.x * 8 + wid;
  const float* hr = h + (size_t)t * DMODEL;

  float acc[NEXP];
  float accs = 0.f;
#pragma unroll
  for (int e = 0; e < NEXP; ++e) acc[e] = 0.f;

  for (int i = lane; i < DMODEL; i += 32) {
    float hv = hr[i];
    const float* gw = gate_w + (size_t)i * NEXP;
#pragma unroll
    for (int e = 0; e < NEXP; ++e) acc[e] += hv * gw[e];
    accs += hv * wsg[i];
  }
#pragma unroll
  for (int off = 16; off > 0; off >>= 1) {
#pragma unroll
    for (int e = 0; e < NEXP; ++e) acc[e] += __shfl_xor(acc[e], off, 32);
    accs += __shfl_xor(accs, off, 32);
  }

  if (lane == 0) {
    float m = acc[0];
#pragma unroll
    for (int e = 1; e < NEXP; ++e) m = fmaxf(m, acc[e]);
    float p[NEXP], s = 0.f;
#pragma unroll
    for (int e = 0; e < NEXP; ++e) { p[e] = __expf(acc[e] - m); s += p[e]; }
    float inv = 1.f / s;
#pragma unroll
    for (int e = 0; e < NEXP; ++e) p[e] *= inv;

    int i0 = 0;
#pragma unroll
    for (int e = 1; e < NEXP; ++e) if (p[e] > p[i0]) i0 = e;
    int i1 = (i0 == 0) ? 1 : 0;
#pragma unroll
    for (int e = 0; e < NEXP; ++e) if (e != i0 && p[e] > p[i1]) i1 = e;

    float wn = 1.f / (p[i0] + p[i1]);
    float* cr = comb + (size_t)t * NEXP;
#pragma unroll
    for (int e = 0; e < NEXP; ++e) cr[e] = 0.f;
    cr[i0] = p[i0] * wn;
    cr[i1] = p[i1] * wn;
    gsh[t] = 1.f / (1.f + __expf(-accs));
  }
}

// ---------------------------------------------------------------------------
// Fragment helper (ISA §7.12.2, wave32):
// 16-bit A/B frag: lane<16 -> K={0..7,16..23}, lane>=16 -> K={8..15,24..31}
// ---------------------------------------------------------------------------
__device__ __forceinline__ v16bf make_frag(const bf16* row, int kb) {
  v8bf lo = *(const v8bf*)(row + kb);
  v8bf hi = *(const v8bf*)(row + 16 + kb);
  return __builtin_shufflevector(lo, hi, 0, 1, 2, 3, 4, 5, 6, 7,
                                 8, 9, 10, 11, 12, 13, 14, 15);
}

// ---------------------------------------------------------------------------
// Fused gate/up GEMM + SiLU epilogue: act = silu(A@Bg) * (A@Bu)  (bf16 out)
// ---------------------------------------------------------------------------
__global__ __launch_bounds__(256)
void gemm_gateup_silu(const bf16* __restrict__ A,
                      const float* __restrict__ Bg,
                      const float* __restrict__ Bu,
                      bf16* __restrict__ act,
                      int M, int K, int N) {
  __shared__ __align__(16) bf16 sA [BM][BK];
  __shared__ __align__(16) bf16 sBg[BN][BK];
  __shared__ __align__(16) bf16 sBu[BN][BK];

  const int tid  = threadIdx.x;
  const int lane = tid & 31;
  const int wid  = tid >> 5;
  const int wm   = wid & 1;        // 2 waves along M (64 rows each)
  const int wn   = wid >> 1;       // 4 waves along N (32 cols each)
  const int m0   = blockIdx.y * BM;
  const int n0   = blockIdx.x * BN;

  const int half = lane >> 4;
  const int l16  = lane & 15;
  const int kb   = half * 8;

  v8f accg[4][2];
  v8f accu[4][2];
#pragma unroll
  for (int mi = 0; mi < 4; ++mi)
#pragma unroll
    for (int ni = 0; ni < 2; ++ni) { accg[mi][ni] = {}; accu[mi][ni] = {}; }

  for (int k0 = 0; k0 < K; k0 += BK) {
    stage_A_tile(A, sA, m0, k0, K, tid);
    stage_B_tile(Bg, sBg, k0, n0, N, tid);
    stage_B_tile(Bu, sBu, k0, n0, N, tid);
    if (k0 + BK < K) {  // global_prefetch_b8 next weight tiles
      __builtin_prefetch(&Bg[(size_t)(k0 + BK + (tid >> 2)) * N + n0 +
                             ((tid & 3) << 5)], 0, 0);
      __builtin_prefetch(&Bu[(size_t)(k0 + BK + (tid >> 2)) * N + n0 +
                             ((tid & 3) << 5)], 0, 0);
    }
    wait_async_lds();
    __syncthreads();

#pragma unroll
    for (int ks = 0; ks < 2; ++ks) {
      v16bf af[4];
#pragma unroll
      for (int mi = 0; mi < 4; ++mi)
        af[mi] = make_frag(&sA[wm * 64 + mi * 16 + l16][ks * 32], kb);
#pragma unroll
      for (int ni = 0; ni < 2; ++ni) {
        int nr = wn * 32 + ni * 16 + l16;
        v16bf bg = make_frag(&sBg[nr][ks * 32], kb);
        v16bf bu = make_frag(&sBu[nr][ks * 32], kb);
#pragma unroll
        for (int mi = 0; mi < 4; ++mi) {
          accg[mi][ni] = __builtin_amdgcn_wmma_f32_16x16x32_bf16(
              false, af[mi], false, bg, (short)0, accg[mi][ni], false, false);
          accu[mi][ni] = __builtin_amdgcn_wmma_f32_16x16x32_bf16(
              false, af[mi], false, bu, (short)0, accu[mi][ni], false, false);
        }
      }
    }
    __syncthreads();
  }

#pragma unroll
  for (int mi = 0; mi < 4; ++mi)
#pragma unroll
    for (int ni = 0; ni < 2; ++ni)
#pragma unroll
      for (int r = 0; r < 8; ++r) {
        int row = m0 + wm * 64 + mi * 16 + half * 8 + r;
        int col = n0 + wn * 32 + ni * 16 + l16;
        float gv = accg[mi][ni][r];
        float uv = accu[mi][ni][r];
        float a  = gv * (1.f / (1.f + __expf(-gv))) * uv;
        act[(size_t)row * N + col] = (bf16)a;
      }
}

// ---------------------------------------------------------------------------
// Down GEMM with per-row scale: out[m,n] (+)= scale[m*sstride]*(A@B)[m,n]
// ---------------------------------------------------------------------------
template <bool ACCUM>
__global__ __launch_bounds__(256)
void gemm_down_scaled(const bf16* __restrict__ A,
                      const float* __restrict__ B,
                      float* __restrict__ out,
                      const float* __restrict__ scale, int sstride,
                      int M, int K, int N) {
  __shared__ __align__(16) bf16 sA[BM][BK];
  __shared__ __align__(16) bf16 sB[BN][BK];

  const int tid  = threadIdx.x;
  const int lane = tid & 31;
  const int wid  = tid >> 5;
  const int wm   = wid & 1;
  const int wn   = wid >> 1;
  const int m0   = blockIdx.y * BM;
  const int n0   = blockIdx.x * BN;

  const int half = lane >> 4;
  const int l16  = lane & 15;
  const int kb   = half * 8;

  v8f acc[4][2];
#pragma unroll
  for (int mi = 0; mi < 4; ++mi)
#pragma unroll
    for (int ni = 0; ni < 2; ++ni) acc[mi][ni] = {};

  for (int k0 = 0; k0 < K; k0 += BK) {
    stage_A_tile(A, sA, m0, k0, K, tid);
    stage_B_tile(B, sB, k0, n0, N, tid);
    if (k0 + BK < K) {
      __builtin_prefetch(&B[(size_t)(k0 + BK + (tid >> 2)) * N + n0 +
                            ((tid & 3) << 5)], 0, 0);
    }
    wait_async_lds();
    __syncthreads();

#pragma unroll
    for (int ks = 0; ks < 2; ++ks) {
      v16bf af[4];
#pragma unroll
      for (int mi = 0; mi < 4; ++mi)
        af[mi] = make_frag(&sA[wm * 64 + mi * 16 + l16][ks * 32], kb);
#pragma unroll
      for (int ni = 0; ni < 2; ++ni) {
        v16bf bf = make_frag(&sB[wn * 32 + ni * 16 + l16][ks * 32], kb);
#pragma unroll
        for (int mi = 0; mi < 4; ++mi)
          acc[mi][ni] = __builtin_amdgcn_wmma_f32_16x16x32_bf16(
              false, af[mi], false, bf, (short)0, acc[mi][ni], false, false);
      }
    }
    __syncthreads();
  }

#pragma unroll
  for (int mi = 0; mi < 4; ++mi)
#pragma unroll
    for (int ni = 0; ni < 2; ++ni)
#pragma unroll
      for (int r = 0; r < 8; ++r) {
        int row = m0 + wm * 64 + mi * 16 + half * 8 + r;
        int col = n0 + wn * 32 + ni * 16 + l16;
        float sc = scale[(size_t)row * sstride];
        float v  = sc * acc[mi][ni][r];
        size_t idx = (size_t)row * N + col;
        if (ACCUM) out[idx] += v; else out[idx] = v;
      }
}

// ---------------------------------------------------------------------------
// Launch: cvt -> router -> shared expert (write) -> 8 experts (accumulate)
// ---------------------------------------------------------------------------
extern "C" void kernel_launch(void* const* d_in, const int* in_sizes, int n_in,
                              void* d_out, int out_size, void* d_ws,
                              size_t ws_size, hipStream_t stream) {
  const float* h       = (const float*)d_in[0];
  const float* gate_w  = (const float*)d_in[1];
  const float* w_gate  = (const float*)d_in[2];
  const float* w_up    = (const float*)d_in[3];
  const float* w_down  = (const float*)d_in[4];
  const float* ws_gate = (const float*)d_in[5];
  const float* ws_up   = (const float*)d_in[6];
  const float* ws_down = (const float*)d_in[7];
  const float* wsg     = (const float*)d_in[8];
  float* out = (float*)d_out;

  char*  ws   = (char*)d_ws;
  size_t off  = 0;
  bf16*  hb   = (bf16*)(ws + off);  off += (size_t)TOKENS * DMODEL * sizeof(bf16);
  bf16*  act  = (bf16*)(ws + off);  off += (size_t)TOKENS * FSH * sizeof(bf16);
  float* comb = (float*)(ws + off); off += (size_t)TOKENS * NEXP * sizeof(float);
  float* gsh  = (float*)(ws + off);

  {
    int n = TOKENS * DMODEL;
    cvt_f32_to_bf16<<<n / (4 * 256), 256, 0, stream>>>(h, hb, n);
  }
  router_kernel<<<TOKENS / 8, 256, 0, stream>>>(h, gate_w, wsg, comb, gsh);

  // shared expert
  gemm_gateup_silu<<<dim3(FSH / BN, TOKENS / BM), 256, 0, stream>>>(
      hb, ws_gate, ws_up, act, TOKENS, DMODEL, FSH);
  gemm_down_scaled<false><<<dim3(DMODEL / BN, TOKENS / BM), 256, 0, stream>>>(
      act, ws_down, out, gsh, 1, TOKENS, FSH, DMODEL);

  // routed experts (zero-weighted dense combine, matches reference)
  for (int e = 0; e < NEXP; ++e) {
    gemm_gateup_silu<<<dim3(FEXP / BN, TOKENS / BM), 256, 0, stream>>>(
        hb, w_gate + (size_t)e * DMODEL * FEXP,
        w_up + (size_t)e * DMODEL * FEXP, act, TOKENS, DMODEL, FEXP);
    gemm_down_scaled<true><<<dim3(DMODEL / BN, TOKENS / BM), 256, 0, stream>>>(
        act, w_down + (size_t)e * FEXP * DMODEL, out, comb + e, NEXP,
        TOKENS, FEXP, DMODEL);
  }
}
